// NoisyTokenChoiceRouter_1967095022051
// MI455X (gfx1250) — matrix operand order repaired and checked by reading, not compiled, not run on hardware
//
#include <hip/hip_runtime.h>
#include <hip/hip_bf16.h>
#include <math.h>

typedef __attribute__((ext_vector_type(16))) __bf16          v16bf;
typedef __attribute__((ext_vector_type(8)))  float           v8f;

#define N_TOK 16384
#define DDIM  2048
#define NEXP  64
#define ROWS_PER_BLOCK 128
#define LDS_STRIDE 65     // 64 + 1 pad -> conflict-free column reads
#define NOISE_STD_INV 64.0f
#define INV_SQRT2 0.70710678118654752f

// ---------------------------------------------------------------------------
// Kernel 0: convert W (f32 [2048,64] row-major) into bf16 WMMA B-fragments.
// Fragment layout (per 32x16 K-tile x N-tile): lane L: N = nt*16 + (L&15),
// K = kt*32 + (L>>4)*8 + (e&7) + (e>>3)*16  for v16bf element e.  One contiguous 32B per lane.
// ---------------------------------------------------------------------------
__global__ void pack_w_bf16(const float* __restrict__ W,
                            __bf16* __restrict__ dst) {
  int tid = blockIdx.x * blockDim.x + threadIdx.x;   // 131072 total
  if (tid >= DDIM * NEXP) return;
  int e    = tid & 15;
  int lane = (tid >> 4) & 31;
  int nt   = (tid >> 9) & 3;
  int kt   = tid >> 11;
  int k = kt * 32 + (lane >> 4) * 8 + (e & 7) + (e >> 3) * 16;
  int n = nt * 16 + (lane & 15);
  dst[tid] = (__bf16)W[(size_t)k * NEXP + n];   // native RNE convert
}

// Kernel 0b: zero the 128 global accumulator floats (importance + load sums).
__global__ void zero_accum(float* __restrict__ acc) {
  acc[threadIdx.x] = 0.0f;
}

// ---------------------------------------------------------------------------
// Kernel 1: WMMA GEMM (logits = x@W + b) + per-token router work.
// 8 waves/block; wave w computes rows [blk*128 + w*16, +16) x all 64 experts.
// ---------------------------------------------------------------------------
__global__ __launch_bounds__(256) void router_kernel(
    const float* __restrict__ x, const __bf16* __restrict__ Wp,
    const float* __restrict__ bias, const float* __restrict__ noise,
    float* __restrict__ out_v, int* __restrict__ out_i,
    float* __restrict__ g_imp, float* __restrict__ g_load)
{
  __shared__ float s_logits[ROWS_PER_BLOCK * LDS_STRIDE];
  __shared__ float s_imp[NEXP];
  __shared__ float s_load[NEXP];

  const int wave  = threadIdx.x >> 5;
  const int lane  = threadIdx.x & 31;
  const int khalf = lane >> 4;       // which K-half this lane supplies
  const int nlo   = lane & 15;       // row within A tile / column within B tile
  const int rowTile = blockIdx.x * ROWS_PER_BLOCK + wave * 16;
  const float* xrow = x + (size_t)(rowTile + nlo) * DDIM;

  v8f acc[4] = {};   // 16x64 logits: 4 N-tiles of 16

  for (int kt = 0; kt < DDIM / 32; ++kt) {
    // ---- A fragment: 16x32 bf16. Lane (khalf,nlo): row M=nlo,
    //      elements 0..7  <- K = k0 + 8*khalf + i
    //      elements 8..15 <- K = k0 + 16 + 8*khalf + i   (per ISA layout table)
    const int k0 = kt * 32 + khalf * 8;
    const float4* p0 = reinterpret_cast<const float4*>(xrow + k0);
    const float4* p1 = reinterpret_cast<const float4*>(xrow + k0 + 16);
    float4 f0 = p0[0], f1 = p0[1], f2 = p1[0], f3 = p1[1];
    v16bf a;
    a[0]  = (__bf16)f0.x; a[1]  = (__bf16)f0.y;
    a[2]  = (__bf16)f0.z; a[3]  = (__bf16)f0.w;
    a[4]  = (__bf16)f1.x; a[5]  = (__bf16)f1.y;
    a[6]  = (__bf16)f1.z; a[7]  = (__bf16)f1.w;
    a[8]  = (__bf16)f2.x; a[9]  = (__bf16)f2.y;
    a[10] = (__bf16)f2.z; a[11] = (__bf16)f2.w;
    a[12] = (__bf16)f3.x; a[13] = (__bf16)f3.y;
    a[14] = (__bf16)f3.z; a[15] = (__bf16)f3.w;

    // ---- B fragments: pre-swizzled, 1 contiguous 32B load per lane per tile
    const v16bf* bp = reinterpret_cast<const v16bf*>(Wp + (size_t)kt * 2048);
#pragma unroll
    for (int nt = 0; nt < 4; ++nt) {
      v16bf bfrag = bp[nt * 32 + lane];
      acc[nt] = __builtin_amdgcn_wmma_f32_16x16x32_bf16(
          false, a, false, bfrag, (short)0, acc[nt], false, false);
    }
  }

  // ---- bias + spill tile to LDS.  C layout: VGPR r -> M = r + 8*khalf,
  //      N = nt*16 + nlo.
#pragma unroll
  for (int nt = 0; nt < 4; ++nt) {
    float bv = bias[nt * 16 + nlo];
#pragma unroll
    for (int r = 0; r < 8; ++r) {
      int m = r + khalf * 8;
      s_logits[(wave * 16 + m) * LDS_STRIDE + nt * 16 + nlo] = acc[nt][r] + bv;
    }
  }
  __syncthreads();
  if (threadIdx.x < NEXP) { s_imp[threadIdx.x] = 0.0f; s_load[threadIdx.x] = 0.0f; }
  __syncthreads();

  // ---- Phase 2: one token per thread (128 tokens/block)
  if (threadIdx.x < ROWS_PER_BLOCK) {
    const int tl  = threadIdx.x;
    const int tok = blockIdx.x * ROWS_PER_BLOCK + tl;
    const float* lrow = &s_logits[tl * LDS_STRIDE];
    const float* nz   = noise + (size_t)tok * NEXP;
    const float noise_std = 1.0f / NOISE_STD_INV;

    float ln[NEXP];
    float v1 = -INFINITY, v2 = -INFINITY;
    int   i1 = 0, i2 = 0;
    float mclean = -INFINITY;
#pragma unroll
    for (int e = 0; e < NEXP; ++e) {
      float l   = lrow[e];
      float lne = fmaf(noise_std, nz[e], l);
      ln[e] = lne;
      mclean = fmaxf(mclean, l);
      if (lne > v1)      { v2 = v1; i2 = i1; v1 = lne; i1 = e; }
      else if (lne > v2) { v2 = lne; i2 = e; }
    }
    float sum_c = 0.0f, sum_n = 0.0f;
#pragma unroll
    for (int e = 0; e < NEXP; ++e) {
      sum_c += __expf(lrow[e] - mclean);
      sum_n += __expf(ln[e] - v1);
    }
    const float thr    = v2;              // 2nd-largest noisy logit
    const float inv_sc = 1.0f / sum_c;
#pragma unroll
    for (int e = 0; e < NEXP; ++e) {
      float l = lrow[e];
      float g = __expf(l - mclean) * inv_sc;          // clean gate
      atomicAdd(&s_imp[e], g);
      float z = (thr - l) * NOISE_STD_INV;
      float p = 0.5f * erfcf(z * INV_SQRT2);          // 1 - Phi(z)
      atomicAdd(&s_load[e], p);
    }
    // top-2 noisy gates, renormalized
    float g1 = 1.0f / sum_n;                          // exp(v1-v1)/sum
    float g2 = __expf(v2 - v1) / sum_n;
    float dn = g1 + g2 + 1e-20f;
    out_v[(size_t)tok * 2 + 0] = g1 / dn;
    out_v[(size_t)tok * 2 + 1] = g2 / dn;
    out_i[(size_t)tok * 2 + 0] = i1;
    out_i[(size_t)tok * 2 + 1] = i2;
  }
  __syncthreads();
  if (threadIdx.x < NEXP) {
    atomicAdd(&g_imp[threadIdx.x],  s_imp[threadIdx.x]);
    atomicAdd(&g_load[threadIdx.x], s_load[threadIdx.x]);
  }
}

// ---------------------------------------------------------------------------
// Kernel 2: aux loss = 0.5*(CV^2(importance) + CV^2(load)), ddof=1 std.
// ---------------------------------------------------------------------------
__global__ void finalize_aux(const float* __restrict__ g_imp,
                             const float* __restrict__ g_load,
                             float* __restrict__ out_aux) {
  if (threadIdx.x == 0 && blockIdx.x == 0) {
    const float invN = 1.0f / (float)N_TOK;
    float mi = 0.0f, mp = 0.0f;
    for (int e = 0; e < NEXP; ++e) { mi += g_imp[e]; mp += g_load[e] * invN; }
    mi *= (1.0f / NEXP); mp *= (1.0f / NEXP);
    float vi = 0.0f, vp = 0.0f;
    for (int e = 0; e < NEXP; ++e) {
      float di = g_imp[e] - mi;          vi += di * di;
      float dp = g_load[e] * invN - mp;  vp += dp * dp;
    }
    vi *= (1.0f / (NEXP - 1)); vp *= (1.0f / (NEXP - 1));
    float li = vi / ((mi + 1e-8f) * (mi + 1e-8f));
    float lp = vp / ((mp + 1e-8f) * (mp + 1e-8f));
    out_aux[0] = 0.5f * (li + lp);
  }
}

extern "C" void kernel_launch(void* const* d_in, const int* in_sizes, int n_in,
                              void* d_out, int out_size, void* d_ws, size_t ws_size,
                              hipStream_t stream) {
  const float* x     = (const float*)d_in[0];   // [16384,2048]
  const float* W     = (const float*)d_in[1];   // [2048,64]
  const float* bias  = (const float*)d_in[2];   // [64]
  const float* noise = (const float*)d_in[3];   // [16384,64]

  __bf16* Wp    = (__bf16*)d_ws;                                  // 256 KB packed bf16 W
  float* g_imp  = (float*)((char*)d_ws + (size_t)DDIM * NEXP * 2);
  float* g_load = g_imp + NEXP;

  float* out_v   = (float*)d_out;                      // [16384,2] top-k gate values
  int*   out_i   = (int*)((float*)d_out + 2 * N_TOK);  // [16384,2] indices (i32 bits)
  float* out_aux = (float*)d_out + 4 * N_TOK;          // [1] aux loss

  pack_w_bf16<<<(DDIM * NEXP) / 256, 256, 0, stream>>>(W, Wp);
  zero_accum<<<1, 2 * NEXP, 0, stream>>>(g_imp);
  router_kernel<<<N_TOK / ROWS_PER_BLOCK, 256, 0, stream>>>(
      x, Wp, bias, noise, out_v, out_i, g_imp, g_load);
  finalize_aux<<<1, 32, 0, stream>>>(g_imp, g_load, out_aux);
}
